// ImageCaptionModel_50148038148714
// MI455X (gfx1250) — compile-verified, matrix operand order checked by reading, not run on hardware
//
#include <hip/hip_runtime.h>
#include <hip/hip_bf16.h>
#include <stdint.h>

// ---------------- problem dimensions ----------------
constexpr int kB  = 128;
constexpr int kT  = 32;
constexpr int kV  = 10000;
constexpr int kH  = 512;
constexpr int kE  = 512;
constexpr int kF  = 2048;
constexpr int kNM = 512;
constexpr int kD0 = kE + kNM + kH;   // 1536
constexpr int kD1 = kH + kH;         // 1024
constexpr int kG  = 4 * kH;          // 2048

typedef __attribute__((ext_vector_type(16))) __bf16   v16bf;
typedef __attribute__((ext_vector_type(8)))  float    v8f;
typedef __attribute__((ext_vector_type(4)))  unsigned u32x4;   // POD vector (not HIP class)

// explicitly global-address-space element types (forces global_load encodings)
typedef __attribute__((address_space(1))) const unsigned short gu16;
typedef __attribute__((address_space(1))) const u32x4          gu4;

struct BF16Frag { u32x4 lo, hi; };

static __device__ __forceinline__ unsigned short f2bf(float f) {
  unsigned u = __builtin_bit_cast(unsigned, f);
  unsigned r = u + 0x7FFFu + ((u >> 16) & 1u);   // round-to-nearest-even
  return (unsigned short)(r >> 16);
}

static __device__ __forceinline__ float sigm(float x) {
  return 1.0f / (1.0f + __expf(-x));
}

// Load one 16x32 bf16 fragment from a lane-adjusted GLOBAL pointer at element offset.
// Pointer already encodes: row = tile_row + (lane&15), koffset = (lane>>4)*8.
// Chunk0 at +eoff, chunk1 at +eoff+16 (per the ISA 16-bit A/B VGPR layout).
static __device__ __forceinline__ v16bf ld_frag(const gu16* __restrict__ p, int eoff) {
  const gu4* q = (const gu4*)(p + eoff);
  BF16Frag fr;
  fr.lo = q[0];
  fr.hi = q[2];        // +32 bytes
  return __builtin_bit_cast(v16bf, fr);
}

#define WMMA_BF16(a, b, c) \
  __builtin_amdgcn_wmma_f32_16x16x32_bf16(false, (a), false, (b), (short)0, (c), false, false)

// ---------------- bf16 WMMA GEMM (ping-pong pipelined) ----------------
// C[M=128, *] = A[128, K](bf16, lda) * Bt[*, K](bf16, ldb)^T + bias.
// Wave computes a 32 x (NT*16) strip: 2 M-tiles x NT N-tiles.
// Main launch: 256 threads (8 waves = 4 M-groups x 2 N-groups), block covers 2*NT*16 cols.
// Tail launch: 128 threads (4 waves), 1 N-group, starting at `nbase`.
// Control flow is scalar-only -> EXEC stays all-1s for WMMA.
// K/32 must be even (true for K in {512, 1024, 1536, 2048}).
template<int NT, bool NTSTORE>
__global__ void gemm_wmma(const unsigned short* __restrict__ A_, int lda,
                          const unsigned short* __restrict__ Bt_, int ldb,
                          float* __restrict__ C, long ldc,
                          const float* __restrict__ bias,
                          int K, int act, int nbase)
{
  const gu16* A  = (const gu16*)A_;
  const gu16* Bt = (const gu16*)Bt_;

  const int lane = threadIdx.x & 31;
  const int wave = threadIdx.x >> 5;
  const int mt   = wave & 3;                 // 4 M-groups of 32 rows (M = 128)
  const int ngs  = wave >> 2;                // 0/1 N-subgroup within block
  const int m0   = mt << 5;
  const int n0   = nbase + (blockIdx.x * 2 + ngs) * (NT * 16);

  const int lrow = lane & 15;
  const int lkof = (lane >> 4) << 3;

  // lane-adjusted base pointers (advance by 64 elements per 2 K-steps)
  const gu16* pA0 = A  + (size_t)(m0 + lrow)      * lda + lkof;
  const gu16* pA1 = A  + (size_t)(m0 + 16 + lrow) * lda + lkof;
  const gu16* pB[NT];
  #pragma unroll
  for (int j = 0; j < NT; ++j)
    pB[j] = Bt + (size_t)(n0 + (j << 4) + lrow) * ldb + lkof;

  v8f acc[2][NT];
  #pragma unroll
  for (int i = 0; i < 2; ++i)
    #pragma unroll
    for (int j = 0; j < NT; ++j)
      acc[i][j] = v8f{0.f, 0.f, 0.f, 0.f, 0.f, 0.f, 0.f, 0.f};

  // ping set (X) holds K-step s, pong set (Y) holds K-step s+1. S = K/32 even.
  v16bf a0X = ld_frag(pA0, 0);
  v16bf a1X = ld_frag(pA1, 0);
  v16bf bX[NT];
  #pragma unroll
  for (int j = 0; j < NT; ++j) bX[j] = ld_frag(pB[j], 0);

  v16bf a0Y, a1Y, bY[NT];

  const int S = K >> 5;
  for (int s = 1; s + 1 < S; s += 2) {
    // 1) issue pong loads (offset +32 elems)
    a0Y = ld_frag(pA0, 32);
    a1Y = ld_frag(pA1, 32);
    #pragma unroll
    for (int j = 0; j < NT; ++j) bY[j] = ld_frag(pB[j], 32);

    // 2) compute on ping (loads issued one full iteration ago)
    #pragma unroll
    for (int j = 0; j < NT; ++j) {
      acc[0][j] = WMMA_BF16(a0X, bX[j], acc[0][j]);
      acc[1][j] = WMMA_BF16(a1X, bX[j], acc[1][j]);
    }

    // 3) advance bases by 2 K-steps
    pA0 += 64; pA1 += 64;
    #pragma unroll
    for (int j = 0; j < NT; ++j) pB[j] += 64;

    // 4) issue next ping loads (new base, offset 0)
    a0X = ld_frag(pA0, 0);
    a1X = ld_frag(pA1, 0);
    #pragma unroll
    for (int j = 0; j < NT; ++j) bX[j] = ld_frag(pB[j], 0);

    // 5) compute on pong
    #pragma unroll
    for (int j = 0; j < NT; ++j) {
      acc[0][j] = WMMA_BF16(a0Y, bY[j], acc[0][j]);
      acc[1][j] = WMMA_BF16(a1Y, bY[j], acc[1][j]);
    }
  }

  // epilogue: ping holds step S-2; load pong for step S-1, compute both
  a0Y = ld_frag(pA0, 32);
  a1Y = ld_frag(pA1, 32);
  #pragma unroll
  for (int j = 0; j < NT; ++j) bY[j] = ld_frag(pB[j], 32);
  #pragma unroll
  for (int j = 0; j < NT; ++j) {
    acc[0][j] = WMMA_BF16(a0X, bX[j], acc[0][j]);
    acc[1][j] = WMMA_BF16(a1X, bX[j], acc[1][j]);
  }
  #pragma unroll
  for (int j = 0; j < NT; ++j) {
    acc[0][j] = WMMA_BF16(a0Y, bY[j], acc[0][j]);
    acc[1][j] = WMMA_BF16(a1Y, bY[j], acc[1][j]);
  }

  const int col_l   = lane & 15;
  const int row_off = (lane >> 4) << 3;      // lanes 16-31 hold M = r+8
  #pragma unroll
  for (int j = 0; j < NT; ++j) {
    const int col = n0 + (j << 4) + col_l;
    const float bv = bias[col];
    #pragma unroll
    for (int i = 0; i < 2; ++i) {
      const int rbase = m0 + (i << 4) + row_off;
      #pragma unroll
      for (int r = 0; r < 8; ++r) {
        float v = acc[i][j][r] + bv;
        if (act) v = (v > 0.f) ? v : 0.01f * v;
        if (NTSTORE)
          __builtin_nontemporal_store(v, &C[(long)(rbase + r) * ldc + col]);
        else
          C[(long)(rbase + r) * ldc + col] = v;
      }
    }
  }
}

// ---------------- data movement / elementwise ----------------

// in [K, N] f32 row-major -> out [N, K] bf16 row-major (weights become [N,K] for B frags)
__global__ void transpose_to_bf16(const float* __restrict__ in,
                                  unsigned short* __restrict__ out, int K, int N) {
  long tid = (long)blockIdx.x * blockDim.x + threadIdx.x;
  if (tid >= (long)K * N) return;
  int k = (int)(tid / N), n = (int)(tid % N);
  out[(long)n * K + k] = f2bf(in[tid]);
}

__global__ void convert_bf16(const float* __restrict__ in,
                             unsigned short* __restrict__ out, long n) {
  long tid = (long)blockIdx.x * blockDim.x + threadIdx.x;
  if (tid < n) out[tid] = f2bf(in[tid]);
}

// f32 [rows, ldin] -> bf16 [rows, ldout] (strided slice pack)
__global__ void slice_to_bf16(const float* __restrict__ in, int ldin,
                              unsigned short* __restrict__ out, int ldout,
                              int rows, int cols) {
  int tid = blockIdx.x * blockDim.x + threadIdx.x;
  if (tid >= rows * cols) return;
  int r = tid / cols, c = tid % cols;
  out[(long)r * ldout + c] = f2bf(in[(long)r * ldin + c]);
}

// gather emb[tok[b][t]] into A0 cols [0, E)
__global__ void gather_embed(const int* __restrict__ tok,
                             const float* __restrict__ emb,
                             unsigned short* __restrict__ A0, int t) {
  int tid = blockIdx.x * blockDim.x + threadIdx.x;   // B*E threads
  if (tid >= kB * kE) return;
  int b = tid >> 9, e = tid & 511;
  int v = tok[b * kT + t];
  A0[(long)b * kD0 + e] = f2bf(emb[(long)v * kE + e]);
}

// unpack current_hidden_state (2,B,2H) into f32 state + bf16 A-matrix slots
__global__ void init_state(const float* __restrict__ chs,
                           float* __restrict__ h0, float* __restrict__ c0,
                           float* __restrict__ h1, float* __restrict__ c1,
                           unsigned short* __restrict__ A0,
                           unsigned short* __restrict__ A1) {
  int tid = blockIdx.x * blockDim.x + threadIdx.x;   // B*H threads
  if (tid >= kB * kH) return;
  int b = tid >> 9, n = tid & 511;
  float vh0 = chs[(long)b * 2 * kH + n];
  float vc0 = chs[(long)b * 2 * kH + kH + n];
  float vh1 = chs[(long)kB * 2 * kH + (long)b * 2 * kH + n];
  float vc1 = chs[(long)kB * 2 * kH + (long)b * 2 * kH + kH + n];
  h0[tid] = vh0; c0[tid] = vc0; h1[tid] = vh1; c1[tid] = vc1;
  A0[(long)b * kD0 + (kE + kNM) + n] = f2bf(vh0);   // h0 slot of x0||h0
  A1[(long)b * kD1 + kH + n]         = f2bf(vh1);   // h1 slot of h0n||h1
}

// z [B, 4H] with split order (i, o, f, c_hat); updates h/c in place,
// writes bf16 h into up to two A-matrix slots.
__global__ void lstm_gates(const float* __restrict__ z,
                           float* __restrict__ h, float* __restrict__ c,
                           unsigned short* __restrict__ hb1, int ld1,
                           unsigned short* __restrict__ hb2, int ld2) {
  int tid = blockIdx.x * blockDim.x + threadIdx.x;   // B*H threads
  if (tid >= kB * kH) return;
  int b = tid >> 9, n = tid & 511;
  const float* zr = z + (long)b * kG;
  float it = sigm(zr[n]);
  float ot = sigm(zr[kH + n]);
  float ft = sigm(zr[2 * kH + n]);
  float ch = tanhf(zr[3 * kH + n]);
  float cn = ft * c[tid] + it * ch;
  float hn = ot * tanhf(cn);
  c[tid] = cn;
  h[tid] = hn;
  unsigned short hb = f2bf(hn);
  hb1[(long)b * ld1 + n] = hb;
  if (hb2) hb2[(long)b * ld2 + n] = hb;
}

// final hidden (2,B,2H) appended after logits in d_out
__global__ void pack_hidden(const float* __restrict__ h0, const float* __restrict__ c0,
                            const float* __restrict__ h1, const float* __restrict__ c1,
                            float* __restrict__ out) {
  int tid = blockIdx.x * blockDim.x + threadIdx.x;   // B*H threads
  if (tid >= kB * kH) return;
  int b = tid >> 9, n = tid & 511;
  out[(long)b * 2 * kH + n]                          = h0[tid];
  out[(long)b * 2 * kH + kH + n]                     = c0[tid];
  out[(long)kB * 2 * kH + (long)b * 2 * kH + n]      = h1[tid];
  out[(long)kB * 2 * kH + (long)b * 2 * kH + kH + n] = c1[tid];
}

// ---------------- driver ----------------
extern "C" void kernel_launch(void* const* d_in, const int* in_sizes, int n_in,
                              void* d_out, int out_size, void* d_ws, size_t ws_size,
                              hipStream_t stream) {
  const float* cnn   = (const float*)d_in[0];
  const int*   tok   = (const int*)d_in[1];
  const float* chs   = (const float*)d_in[2];
  const float* emb   = (const float*)d_in[3];
  const float* W_in  = (const float*)d_in[4];
  const float* b_in  = (const float*)d_in[5];
  const float* W0    = (const float*)d_in[6];
  const float* b0    = (const float*)d_in[7];
  const float* W1    = (const float*)d_in[8];
  const float* b1    = (const float*)d_in[9];
  const float* W_out = (const float*)d_in[10];
  const float* b_out = (const float*)d_in[11];
  float* out = (float*)d_out;

  char* w = (char*)d_ws;
  auto take = [&](size_t bytes) -> char* {
    char* p = w; w += (bytes + 255) & ~(size_t)255; return p;
  };
  unsigned short* Wt_in  = (unsigned short*)take((size_t)kNM * kF * 2);
  unsigned short* Wt0    = (unsigned short*)take((size_t)kG * kD0 * 2);
  unsigned short* Wt1    = (unsigned short*)take((size_t)kG * kD1 * 2);
  unsigned short* Wt_out = (unsigned short*)take((size_t)kV * kH * 2);
  unsigned short* Acnn   = (unsigned short*)take((size_t)kB * kF * 2);
  unsigned short* A0     = (unsigned short*)take((size_t)kB * kD0 * 2);
  unsigned short* A1     = (unsigned short*)take((size_t)kB * kD1 * 2);
  float* proc_f = (float*)take((size_t)kB * kNM * 4);
  float* z0     = (float*)take((size_t)kB * kG * 4);
  float* z1     = (float*)take((size_t)kB * kG * 4);
  float* h0     = (float*)take((size_t)kB * kH * 4);
  float* c0     = (float*)take((size_t)kB * kH * 4);
  float* h1     = (float*)take((size_t)kB * kH * 4);
  float* c1     = (float*)take((size_t)kB * kH * 4);

  auto cdiv = [](long a, long b) { return (a + b - 1) / b; };
  const int TB = 256;

  // weight / input conversion (bf16, B matrices transposed to [N,K])
  transpose_to_bf16<<<cdiv((long)kF * kNM, TB), TB, 0, stream>>>(W_in, Wt_in, kF, kNM);
  transpose_to_bf16<<<cdiv((long)kD0 * kG, TB), TB, 0, stream>>>(W0, Wt0, kD0, kG);
  transpose_to_bf16<<<cdiv((long)kD1 * kG, TB), TB, 0, stream>>>(W1, Wt1, kD1, kG);
  transpose_to_bf16<<<cdiv((long)kH * kV, TB), TB, 0, stream>>>(W_out, Wt_out, kH, kV);
  convert_bf16<<<cdiv((long)kB * kF, TB), TB, 0, stream>>>(cnn, Acnn, (long)kB * kF);

  // proc = leaky_relu(cnn @ W_in + b_in)   [128, 512]; 512 cols / 128 per block = 4 blocks
  gemm_wmma<4, false><<<4, 256, 0, stream>>>(Acnn, kF, Wt_in, kF,
                                             proc_f, kNM, b_in, kF, 1, 0);
  // pack proc into A0 cols [E, E+NM)
  slice_to_bf16<<<cdiv((long)kB * kNM, TB), TB, 0, stream>>>(proc_f, kNM,
                                                            A0 + kE, kD0, kB, kNM);
  // initial hidden state -> f32 state + bf16 slots of A0/A1
  init_state<<<cdiv((long)kB * kH, TB), TB, 0, stream>>>(chs, h0, c0, h1, c1, A0, A1);

  const int eltBlocks = (int)cdiv((long)kB * kH, TB);
  for (int t = 0; t < kT; ++t) {
    // A0 cols [0,E) = emb[tok[:, t]]
    gather_embed<<<cdiv((long)kB * kE, TB), TB, 0, stream>>>(tok, emb, A0, t);
    // z0 = [emb_t, proc, h0] @ W0 + b0   [128, 2048]; 2048/128 = 16 blocks
    gemm_wmma<4, false><<<16, 256, 0, stream>>>(A0, kD0, Wt0, kD0,
                                                z0, kG, b0, kD0, 0, 0);
    // layer-0 gates: h0n -> A0 h0-slot (next step) and A1 cols [0,H) (this step)
    lstm_gates<<<eltBlocks, TB, 0, stream>>>(z0, h0, c0,
                                             A0 + (kE + kNM), kD0, A1, kD1);
    // z1 = [h0n, h1] @ W1 + b1           [128, 2048]
    gemm_wmma<4, false><<<16, 256, 0, stream>>>(A1, kD1, Wt1, kD1,
                                                z1, kG, b1, kD1, 0, 0);
    // layer-1 gates: h1n -> A1 cols [H, 2H)
    lstm_gates<<<eltBlocks, TB, 0, stream>>>(z1, h1, c1,
                                             A1 + kH, kD1, (unsigned short*)nullptr, 0);
    // logits_t = h1n @ W_out + b_out  [128, 10000] -> streamed (non-temporal) stores
    float* Ct = out + (long)t * kV;
    gemm_wmma<4, true><<<78, 256, 0, stream>>>(A1 + kH, kD1, Wt_out, kH,
                                               Ct, (long)kT * kV, b_out, kH, 0, 0);
    // tail: cols [9984, 10000)
    gemm_wmma<1, true><<<1, 128, 0, stream>>>(A1 + kH, kD1, Wt_out, kH,
                                              Ct, (long)kT * kV, b_out, kH, 0, 9984);
  }

  pack_hidden<<<eltBlocks, TB, 0, stream>>>(h0, c0, h1, c1,
                                            out + (long)kB * kT * kV);
}
